// Model_39676907883674
// MI455X (gfx1250) — matrix operand order repaired and checked by reading, not compiled, not run on hardware
//
#include <hip/hip_runtime.h>

typedef _Float16 v16h __attribute__((ext_vector_type(16)));
typedef _Float16 v8h  __attribute__((ext_vector_type(8)));
typedef _Float16 v2h  __attribute__((ext_vector_type(2)));
typedef float    v8f  __attribute__((ext_vector_type(8)));
typedef float    v4f  __attribute__((ext_vector_type(4)));

#define S_LEN 2048
#define D_DIM 64
#define KB    64      // key block (two f16 WMMA K=32 slabs)
#define NW    8       // waves per workgroup (wave32)

static __device__ __forceinline__ v16h cat8(v8h lo, v8h hi) {
  v16h r;
#pragma unroll
  for (int i = 0; i < 8; ++i) { r[i] = lo[i]; r[i + 8] = hi[i]; }
  return r;
}

static __device__ __forceinline__ v8f wmma16(v16h a, v16h b, v8f c) {
  return __builtin_amdgcn_wmma_f32_16x16x32_f16(false, a, false, b, (short)0, c,
                                                false, false);
}

#if __has_builtin(__builtin_amdgcn_exp2f)
#define EXP2F(x) __builtin_amdgcn_exp2f(x)
#else
#define EXP2F(x) exp2f(x)
#endif

// ---- xor-reductions within each 16-lane half, on the VALU pipe -------------
#if __has_builtin(__builtin_amdgcn_permlane16)
static __device__ __forceinline__ float plx(float v, unsigned lo, unsigned hi) {
  int s = __builtin_bit_cast(int, v);
  int r = __builtin_amdgcn_permlane16(s, s, lo, hi, false, false);
  return __builtin_bit_cast(float, r);
}
static __device__ __forceinline__ float red_max16(float v) {
  v = fmaxf(v, plx(v, 0x67452301u, 0xEFCDAB89u));   // xor 1
  v = fmaxf(v, plx(v, 0x54761032u, 0xDCFE98BAu));   // xor 2
  v = fmaxf(v, plx(v, 0x32107654u, 0xBA98FEDCu));   // xor 4
  v = fmaxf(v, plx(v, 0xFEDCBA98u, 0x76543210u));   // xor 8
  return v;
}
static __device__ __forceinline__ float red_sum16(float v) {
  v += plx(v, 0x67452301u, 0xEFCDAB89u);
  v += plx(v, 0x54761032u, 0xDCFE98BAu);
  v += plx(v, 0x32107654u, 0xBA98FEDCu);
  v += plx(v, 0xFEDCBA98u, 0x76543210u);
  return v;
}
#else
static __device__ __forceinline__ float red_max16(float v) {
  v = fmaxf(v, __shfl_xor(v, 1, 32));
  v = fmaxf(v, __shfl_xor(v, 2, 32));
  v = fmaxf(v, __shfl_xor(v, 4, 32));
  v = fmaxf(v, __shfl_xor(v, 8, 32));
  return v;
}
static __device__ __forceinline__ float red_sum16(float v) {
  v += __shfl_xor(v, 1, 32);
  v += __shfl_xor(v, 2, 32);
  v += __shfl_xor(v, 4, 32);
  v += __shfl_xor(v, 8, 32);
  return v;
}
#endif

__global__ __launch_bounds__(256)
void fa2_dropout_attn(const float* __restrict__ Q, const float* __restrict__ K,
                      const float* __restrict__ V, const float* __restrict__ SF,
                      const float* __restrict__ M, float* __restrict__ O) {
  __shared__ __align__(32) _Float16 Kt[KB * D_DIM];   // [key][d]          8 KB
  __shared__ __align__(32) _Float16 Vt[D_DIM * KB];   // [d][key] (transp) 8 KB
  __shared__ __align__(32) _Float16 Pl[NW][16 * KB];  // per-wave P       16 KB

  const int lane = threadIdx.x & 31;
  const int wave = threadIdx.x >> 5;
  const int hg   = lane >> 4;        // 0: lanes 0-15, 1: lanes 16-31
  const int ln   = lane & 15;
  const int b    = blockIdx.x >> 4;  // 16 q-blocks per batch
  const int q0   = (blockIdx.x & 15) * (NW * 16) + wave * 16;
  // fold log2(e) into the scale: all scores/maxes tracked in log2 domain
  const float scale2 = SF[0] * 1.4426950408889634f;

  // ---- Q A-fragments: lane L: M=L%16; halves0-7 = K 8*hg+0..7, halves8-15 = K 16+8*hg+0..7
  v16h a0, a1;
  {
    const float* qp = Q + ((size_t)b * S_LEN + q0 + ln) * D_DIM;
    const int o = 8 * hg;
    v4f f0 = *(const v4f*)(qp + o);      v4f f1 = *(const v4f*)(qp + o + 4);
    v4f f2 = *(const v4f*)(qp + o + 16); v4f f3 = *(const v4f*)(qp + o + 20);
    v4f g0 = *(const v4f*)(qp + o + 32); v4f g1 = *(const v4f*)(qp + o + 36);
    v4f g2 = *(const v4f*)(qp + o + 48); v4f g3 = *(const v4f*)(qp + o + 52);
#pragma unroll
    for (int i = 0; i < 4; ++i) {
      a0[i]      = (_Float16)f0[i]; a0[i + 4]  = (_Float16)f1[i];
      a0[i + 8]  = (_Float16)f2[i]; a0[i + 12] = (_Float16)f3[i];
      a1[i]      = (_Float16)g0[i]; a1[i + 4]  = (_Float16)g1[i];
      a1[i + 8]  = (_Float16)g2[i]; a1[i + 12] = (_Float16)g3[i];
    }
  }

  float m_r[8], l_r[8], alpha[8];
  v8f acc[4];
#pragma unroll
  for (int r = 0; r < 8; ++r) { m_r[r] = -3.0e38f; l_r[r] = 0.0f; }
#pragma unroll
  for (int c = 0; c < 4; ++c) {
#pragma unroll
    for (int r = 0; r < 8; ++r) acc[c][r] = 0.0f;
  }

  const float* kbase = K + (size_t)b * S_LEN * D_DIM;
  const float* vbase = V + (size_t)b * S_LEN * D_DIM;
  const float* mbase = M + (size_t)b * S_LEN * S_LEN;

  for (int k0 = 0; k0 < S_LEN; k0 += KB) {
    __syncthreads();   // protect Kt/Vt from previous-iteration readers

    // ---- mask loads hoisted to the top: the dominant HBM stream gets the
    //      whole staging + QK-wmma phase as latency cover
    float mk[8][4];
#pragma unroll
    for (int r = 0; r < 8; ++r) {
      const float* mrow = mbase + (size_t)(q0 + r + 8 * hg) * S_LEN + k0;
#pragma unroll
      for (int g = 0; g < 4; ++g) mk[r][g] = mrow[ln + 16 * g];
    }

    // ---- prefetch next K/V tiles (waves 0-3: K, 4-7: V) + next mask block
    if (k0 + KB < S_LEN) {
      const int wq = wave & 3;
      const float* tb = (wave < 4) ? kbase : vbase;
      __builtin_prefetch(tb + (size_t)(k0 + KB) * D_DIM + wq * 1024 + lane * 32, 0, 0);
      __builtin_prefetch(mbase + (size_t)(q0 + ln) * S_LEN + (k0 + KB) + hg * 32, 0, 0);
    }

    // ---- stage K tile [key][d] as f16, packed b128 stores
    {
      const int tr = (int)threadIdx.x >> 2;          // key row 0..63
      const int tc = ((int)threadIdx.x & 3) << 4;    // 0,16,32,48
      const float* src = kbase + (size_t)(k0 + tr) * D_DIM + tc;
      v4f f0 = *(const v4f*)(src);     v4f f1 = *(const v4f*)(src + 4);
      v4f f2 = *(const v4f*)(src + 8); v4f f3 = *(const v4f*)(src + 12);
      v8h h0, h1;
#pragma unroll
      for (int i = 0; i < 4; ++i) {
        h0[i] = (_Float16)f0[i]; h0[i + 4] = (_Float16)f1[i];
        h1[i] = (_Float16)f2[i]; h1[i + 4] = (_Float16)f3[i];
      }
      *(v8h*)&Kt[tr * D_DIM + tc]     = h0;
      *(v8h*)&Kt[tr * D_DIM + tc + 8] = h1;
    }
    // ---- stage V transposed [d][key], key-pairs packed into b32 stores
    {
      const float* vp = vbase + (size_t)k0 * D_DIM;
#pragma unroll
      for (int j = 0; j < 8; ++j) {
        int i = (int)threadIdx.x + (j << 8);
        int d = i & 63;
        int k = (i >> 6) << 1;   // even key
        v2h pk;
        pk[0] = (_Float16)vp[(size_t)k * D_DIM + d];
        pk[1] = (_Float16)vp[(size_t)(k + 1) * D_DIM + d];
        *(v2h*)&Vt[d * KB + k] = pk;
      }
    }
    __syncthreads();

    // ---- Q·K^T : 4 key-groups x (2 d-chunks) = 8 wmma
    v8f c[4];
#pragma unroll
    for (int g = 0; g < 4; ++g) {
      const _Float16* kr = &Kt[(g * 16 + ln) * D_DIM];
      v16h b0 = cat8(*(const v8h*)&kr[8 * hg],      *(const v8h*)&kr[16 + 8 * hg]);
      v16h b1 = cat8(*(const v8h*)&kr[32 + 8 * hg], *(const v8h*)&kr[48 + 8 * hg]);
      v8f z;
#pragma unroll
      for (int r = 0; r < 8; ++r) z[r] = 0.0f;
      z    = wmma16(a0, b0, z);
      c[g] = wmma16(a1, b1, z);
    }

    // ---- online softmax (log2 domain) + dropout (mask on P only)
#pragma unroll
    for (int r = 0; r < 8; ++r) {
      float s0 = c[0][r] * scale2, s1 = c[1][r] * scale2;
      float s2 = c[2][r] * scale2, s3 = c[3][r] * scale2;
      float mx = red_max16(fmaxf(fmaxf(s0, s1), fmaxf(s2, s3)));
      float mnew = fmaxf(m_r[r], mx);
      float al = EXP2F(m_r[r] - mnew);
      float p0 = EXP2F(s0 - mnew), p1 = EXP2F(s1 - mnew);
      float p2 = EXP2F(s2 - mnew), p3 = EXP2F(s3 - mnew);
      float rs = red_sum16((p0 + p1) + (p2 + p3));
      l_r[r] = l_r[r] * al + rs;
      m_r[r] = mnew;
      alpha[r] = al;

      const int row = r + 8 * hg;
      p0 = (mk[r][0] >= 0.5f) ? p0 * 2.0f : 0.0f;   // inverted dropout, 1/(1-0.5)
      p1 = (mk[r][1] >= 0.5f) ? p1 * 2.0f : 0.0f;
      p2 = (mk[r][2] >= 0.5f) ? p2 * 2.0f : 0.0f;
      p3 = (mk[r][3] >= 0.5f) ? p3 * 2.0f : 0.0f;
      _Float16* pr = &Pl[wave][row * KB];
      pr[ln]      = (_Float16)p0;
      pr[ln + 16] = (_Float16)p1;
      pr[ln + 32] = (_Float16)p2;
      pr[ln + 48] = (_Float16)p3;
    }

    // intra-wave cross-lane handoff through per-wave LDS region
    asm volatile("s_wait_dscnt 0" ::: "memory");

    // P as A-frags: keys 0..31 and 32..63
    v16h ap0 = cat8(*(const v8h*)&Pl[wave][ln * KB + 8 * hg],
                    *(const v8h*)&Pl[wave][ln * KB + 16 + 8 * hg]);
    v16h ap1 = cat8(*(const v8h*)&Pl[wave][ln * KB + 32 + 8 * hg],
                    *(const v8h*)&Pl[wave][ln * KB + 48 + 8 * hg]);

    // ---- P·V : 4 d-chunks x 2 key-slabs = 8 wmma
#pragma unroll
    for (int ch = 0; ch < 4; ++ch) {
#pragma unroll
      for (int r = 0; r < 8; ++r) acc[ch][r] *= alpha[r];
      const _Float16* vr = &Vt[(ch * 16 + ln) * KB];
      v16h bv0 = cat8(*(const v8h*)&vr[8 * hg],      *(const v8h*)&vr[16 + 8 * hg]);
      v16h bv1 = cat8(*(const v8h*)&vr[32 + 8 * hg], *(const v8h*)&vr[48 + 8 * hg]);
      acc[ch] = wmma16(ap0, bv0, acc[ch]);
      acc[ch] = wmma16(ap1, bv1, acc[ch]);
    }
  }

  // ---- epilogue: divide by full softmax denominator, store
  float* obase = O + ((size_t)b * S_LEN + q0) * D_DIM;
#pragma unroll
  for (int r = 0; r < 8; ++r) {
    const int row = r + 8 * hg;
    float inv = 1.0f / l_r[r];
#pragma unroll
    for (int ch = 0; ch < 4; ++ch) {
      obase[(size_t)row * D_DIM + ch * 16 + ln] = acc[ch][r] * inv;
    }
  }
}

extern "C" void kernel_launch(void* const* d_in, const int* in_sizes, int n_in,
                              void* d_out, int out_size, void* d_ws, size_t ws_size,
                              hipStream_t stream) {
  const float* Q  = (const float*)d_in[0];
  const float* K  = (const float*)d_in[1];
  const float* V  = (const float*)d_in[2];
  const float* SF = (const float*)d_in[3];
  const float* M  = (const float*)d_in[4];
  float* O = (float*)d_out;
  (void)in_sizes; (void)n_in; (void)out_size; (void)d_ws; (void)ws_size;

  dim3 grid(64 * (S_LEN / (NW * 16)));   // 1024 workgroups
  dim3 block(256);                       // 8 wave32 waves
  hipLaunchKernelGGL(fa2_dropout_attn, grid, block, 0, stream, Q, K, V, SF, M, O);
}
